// TFJSP_Decoder_DHJS_Base_21414706937947
// MI455X (gfx1250) — matrix operand order repaired and checked by reading, not compiled, not run on hardware
//
#include <hip/hip_runtime.h>
#include <hip/hip_bf16.h>
#include <math.h>

// ---------------------------------------------------------------------------
// TFJSP decoder for MI455X (gfx1250, wave32, WMMA).
// Single query per batch element => fold K/V projections into the query side
// (rank-16 trick): HBM-bound on the embedding tensor instead of 344 GFLOP.
// Dense algebra via one bf16 WMMA GEMM kernel (4 wmma / k-step, b128 frags).
// Attention tile staged in LDS via global_load_async_to_lds_b128 (ASYNCcnt).
// ---------------------------------------------------------------------------

typedef unsigned short u16;
typedef __bf16 v16bf __attribute__((ext_vector_type(16)));
typedef u16   v16us __attribute__((ext_vector_type(16)));
typedef float v8f   __attribute__((ext_vector_type(8)));

#define NEG_INF (-__builtin_inff())

__device__ inline u16 f2bf(float f) {
  unsigned u = __builtin_bit_cast(unsigned, f);
  unsigned r = u + 0x7FFFu + ((u >> 16) & 1u);
  return (u16)(r >> 16);
}
__device__ inline float bfl(unsigned w) { return __builtin_bit_cast(float, w << 16); }
__device__ inline float bfh(unsigned w) { return __builtin_bit_cast(float, w & 0xFFFF0000u); }
__device__ inline unsigned packbf2(float a, float b) {
  return (unsigned)f2bf(a) | ((unsigned)f2bf(b) << 16);
}

struct U4x2 { uint4 a, b; };
// load a 16-element bf16 fragment from two contiguous 16B chunks
__device__ inline v16bf ldfrag(const u16* p0, const u16* p1) {
  U4x2 t;
  t.a = *(const uint4*)p0;
  t.b = *(const uint4*)p1;
  return __builtin_bit_cast(v16bf, t);
}

// ---------------------------------------------------------------------------
// Weight conversion fp32 -> bf16, optionally transposed (to make the GEMM
// B-operand K-contiguous so fragments load as b128).
// ---------------------------------------------------------------------------
struct ConvArgs {
  const float* src[17];
  u16* dst[17];
  int rows[17];
  int cols[17];
  int trans[17];
};

__global__ void convert_weights(ConvArgs a) {
  int s = blockIdx.y;
  int i = blockIdx.x * 256 + threadIdx.x;
  int cols = a.cols[s];
  int total = a.rows[s] * cols;
  if (i < total) {
    float v = a.src[s][i];
    int r = i / cols, c = i % cols;
    long di = a.trans[s] ? ((long)c * a.rows[s] + r) : (long)i;
    a.dst[s][di] = f2bf(v);
  }
}

// ---------------------------------------------------------------------------
// Mean over nodes + bf16 conversion of embeddings; builds ctxA = [mean, prev]
// ---------------------------------------------------------------------------
__global__ void mean_convert(const float* __restrict__ emb,
                             const float* __restrict__ prev,
                             u16* __restrict__ embB,
                             u16* __restrict__ ctxA) {
  int b = blockIdx.x;
  int e = threadIdx.x;  // 512 threads
  const float* base = emb + ((long)b * 160) * 512 + e;
  float s = 0.f;
  for (int n = 0; n < 160; n++) {
    float v = base[(long)n * 512];
    s += v;
    embB[((long)b * 160 + n) * 512 + e] = f2bf(v);
  }
  ctxA[(long)b * 2048 + e] = f2bf(s * (1.0f / 160.0f));
  for (int i = e; i < 1536; i += 512)
    ctxA[(long)b * 2048 + 512 + i] = f2bf(prev[(long)b * 1536 + i]);
}

// ---------------------------------------------------------------------------
// ninf masks
// ---------------------------------------------------------------------------
__global__ void ninf_precompute(const unsigned char* __restrict__ mask,
                                const unsigned char* __restrict__ maskVeh,
                                float* __restrict__ ninfNodes,
                                float* __restrict__ ninfOpe,
                                float* __restrict__ ninfVeh) {
  int b = blockIdx.x, t = threadIdx.x;  // 256 threads
  const unsigned char* mb = mask + (long)b * 100 * 30;
  for (int j = t; j < 100; j += 256) {
    int any = 0;
    for (int m = 0; m < 30; m++) any |= mb[j * 30 + m];
    float v = any ? 0.f : NEG_INF;
    ninfOpe[(long)b * 100 + j] = v;
    ninfNodes[(long)b * 160 + j] = v;
  }
  for (int m = t; m < 30; m += 256) {
    int any = 0;
    for (int j = 0; j < 100; j++) any |= mb[j * 30 + m];
    ninfNodes[(long)b * 160 + 100 + m] = any ? 0.f : NEG_INF;
    float v = maskVeh[(long)b * 30 + m] ? 0.f : NEG_INF;
    ninfVeh[(long)b * 30 + m] = v;
    ninfNodes[(long)b * 160 + 130 + m] = v;
  }
}

__global__ void finish_init(unsigned char* fin, int B) {
  int i = blockIdx.x * 256 + threadIdx.x;
  if (i < B) fin[i] = 1;
}
__global__ void finish_scatter(const int* __restrict__ idx, unsigned char* fin, int cnt) {
  int i = blockIdx.x * 256 + threadIdx.x;
  if (i < cnt) fin[idx[i]] = 0;
}

__global__ void ninf_ma_sel(const unsigned char* __restrict__ mask,
                            const int* __restrict__ selJob,
                            float* __restrict__ ninfMa, int B) {
  int i = blockIdx.x * 256 + threadIdx.x;
  if (i < B * 30) {
    int b = i / 30, m = i % 30;
    ninfMa[i] = mask[((long)b * 100 + selJob[b]) * 30 + m] ? 0.f : NEG_INF;
  }
}

// ---------------------------------------------------------------------------
// bf16 WMMA GEMM:  C[z][m][n] = alpha * sum_k A[z][m][k]*B[z][n][k] (+bias[n])
// A: a[m*lda + k] (k contiguous).  B: b[n*ldbn + k] (k contiguous).
// 256 threads = 8 waves stacked along M; block tile 256(M) x 32(N); each wave
// computes 2x2 16x16 C tiles -> 4 v_wmma_f32_16x16x32_bf16 per k-step.
// Fragment layouts per CDNA5 ISA 7.12.2 (wave32): each fragment = two
// contiguous 16B chunks per lane -> global_load_b128.
// ---------------------------------------------------------------------------
struct GemmArgs {
  const u16* A; long lda; long aBatch;
  const u16* B; long ldbn; long bBatch;
  u16* Cb; float* Cf; long ldc; long cBatch;
  const float* bias;
  int M, N, K;
  float alpha;
};

__global__ void __launch_bounds__(256) wmma_gemm(GemmArgs g) {
  const int z = blockIdx.z;
  const u16* A = g.A + (long)z * g.aBatch;
  const u16* Bm = g.B + (long)z * g.bBatch;
  const int wave = threadIdx.x >> 5;
  const int lane = threadIdx.x & 31;
  const int m0 = blockIdx.x * 256 + wave * 32;  // two M tiles: m0, m0+16
  const int n0 = blockIdx.y * 32;               // two N tiles: n0, n0+16
  const int row = lane & 15;
  const int half = lane >> 4;

  v8f acc00 = {}, acc01 = {}, acc10 = {}, acc11 = {};
  // A lane row pointers (rows m0+row and m0+16+row)
  const u16* ar0 = A + (long)(m0 + row) * g.lda;
  const u16* ar1 = A + (long)(m0 + 16 + row) * g.lda;
  // B lane column pointers (cols n0+row and n0+16+row)
  const u16* br0 = Bm + (long)(n0 + row) * g.ldbn;
  const u16* br1 = Bm + (long)(n0 + 16 + row) * g.ldbn;

  for (int k0 = 0; k0 < g.K; k0 += 32) {
    // A frag: elems [kb, kb+8) and [kb+16, kb+24), kb = k0 + half*8
    const int kb = k0 + half * 8;
    v16bf a0 = ldfrag(ar0 + kb, ar0 + kb + 16);
    v16bf a1 = ldfrag(ar1 + kb, ar1 + kb + 16);
    // B frag: elems [kc, kc+8) and [kc+16, kc+24), kc = k0 + half*8
    const int kc = k0 + half * 8;
    v16bf b0 = ldfrag(br0 + kc, br0 + kc + 16);
    v16bf b1 = ldfrag(br1 + kc, br1 + kc + 16);
    acc00 = __builtin_amdgcn_wmma_f32_16x16x32_bf16(false, a0, false, b0, (short)0, acc00, false, false);
    acc01 = __builtin_amdgcn_wmma_f32_16x16x32_bf16(false, a0, false, b1, (short)0, acc01, false, false);
    acc10 = __builtin_amdgcn_wmma_f32_16x16x32_bf16(false, a1, false, b0, (short)0, acc10, false, false);
    acc11 = __builtin_amdgcn_wmma_f32_16x16x32_bf16(false, a1, false, b1, (short)0, acc11, false, false);
  }

  // epilogue: C rows m0 + half*8 + i (tile0), +16 (tile1); cols n0+row, +16
  const int c0 = n0 + row, c1 = n0 + 16 + row;
  const float bv0 = g.bias ? g.bias[c0] : 0.f;
  const float bv1 = g.bias ? g.bias[c1] : 0.f;
  const long r0 = m0 + half * 8;
  const long r1 = r0 + 16;
  if (g.Cf) {
    float* cf = g.Cf + (long)z * g.cBatch;
#pragma unroll
    for (int i = 0; i < 8; i++) {
      cf[(r0 + i) * g.ldc + c0] = acc00[i] * g.alpha + bv0;
      cf[(r0 + i) * g.ldc + c1] = acc01[i] * g.alpha + bv1;
      cf[(r1 + i) * g.ldc + c0] = acc10[i] * g.alpha + bv0;
      cf[(r1 + i) * g.ldc + c1] = acc11[i] * g.alpha + bv1;
    }
  }
  if (g.Cb) {
    u16* cb = g.Cb + (long)z * g.cBatch;
#pragma unroll
    for (int i = 0; i < 8; i++) {
      cb[(r0 + i) * g.ldc + c0] = f2bf(acc00[i] * g.alpha + bv0);
      cb[(r0 + i) * g.ldc + c1] = f2bf(acc01[i] * g.alpha + bv1);
      cb[(r1 + i) * g.ldc + c0] = f2bf(acc10[i] * g.alpha + bv0);
      cb[(r1 + i) * g.ldc + c1] = f2bf(acc11[i] * g.alpha + bv1);
    }
  }
}

// ---------------------------------------------------------------------------
// Fused rank-16 attention per batch element (memory-bound part).
// Embed slice (n x 512 bf16, <=160KB) staged into LDS with the CDNA5 async
// path (global_load_async_to_lds_b128, tracked by ASYNCcnt).
// ---------------------------------------------------------------------------
#define SSTRIDE 164  // pad so h*SSTRIDE hits distinct LDS banks

__device__ inline void async_copy16(unsigned ldsOff, const void* gaddr) {
  asm volatile("global_load_async_to_lds_b128 %0, %1, off"
               :: "v"(ldsOff), "v"(gaddr) : "memory");
}

__global__ void __launch_bounds__(512) attn_fused(
    const u16* __restrict__ embB, int rowOff, int n,
    const u16* __restrict__ Rm, const float* __restrict__ ninf,
    u16* __restrict__ U) {
  extern __shared__ u16 sm[];
  u16* sEmb = sm;                                  // n*512 bf16
  u16* sR   = sm + n * 512;                        // 16 x 520 (padded)
  float* sS = (float*)(sm + n * 512 + 16 * 520);   // 16 x SSTRIDE

  const int b = blockIdx.x, tid = threadIdx.x;  // 512 threads

  // async-copy embed slice into LDS (16B per lane per issue).
  // LDS aperture: low 32 bits of a flat shared address are the LDS offset.
  {
    const char* gsrc = (const char*)(embB + ((long)b * 160 + rowOff) * 512);
    unsigned ldsBase = (unsigned)(size_t)(void*)sEmb;
    int chunks = n * 64;  // n*1024B / 16B
    for (int i = tid; i < chunks; i += 512)
      async_copy16(ldsBase + (unsigned)i * 16u, gsrc + (long)i * 16);
  }
  // R tile via regular loads (16KB)
  {
    const unsigned* rsrc = (const unsigned*)(Rm + (long)b * 8192);
    unsigned* r32 = (unsigned*)sR;
    for (int i = tid; i < 16 * 256; i += 512) {
      int h = i >> 8, e2 = i & 255;
      r32[h * 260 + e2] = rsrc[i];
    }
  }
  asm volatile("s_wait_asynccnt 0x0" ::: "memory");
  __syncthreads();

  // scores: S[h,n] = embed[n,:] . R[h,:] + ninf[n]
  for (int i = tid; i < 16 * n; i += 512) {
    int h = i & 15, nn = i >> 4;
    const unsigned* er = (const unsigned*)sEmb + nn * 256;
    const unsigned* rr = (const unsigned*)sR + h * 260;
    float acc = 0.f;
    for (int e2 = 0; e2 < 256; e2++) {
      unsigned ew = er[e2], rw = rr[e2];
      acc += bfl(ew) * bfl(rw) + bfh(ew) * bfh(rw);
    }
    sS[h * SSTRIDE + nn] = acc + ninf[(long)b * n + nn];
  }
  __syncthreads();

  // softmax: wave w handles head h=w (wave32 shuffles)
  {
    int h = tid >> 5, l = tid & 31;
    float m = NEG_INF;
    for (int nn = l; nn < n; nn += 32) m = fmaxf(m, sS[h * SSTRIDE + nn]);
    for (int o = 16; o; o >>= 1) m = fmaxf(m, __shfl_xor(m, o, 32));
    float s = 0.f;
    for (int nn = l; nn < n; nn += 32) {
      float e = __expf(sS[h * SSTRIDE + nn] - m);
      sS[h * SSTRIDE + nn] = e;
      s += e;
    }
    for (int o = 16; o; o >>= 1) s += __shfl_xor(s, o, 32);
    float inv = 1.0f / s;
    for (int nn = l; nn < n; nn += 32) sS[h * SSTRIDE + nn] *= inv;
  }
  __syncthreads();

  // U[b,h,e] = sum_n a[h,n]*embed[n,e]
  unsigned* U32 = (unsigned*)(U + (long)b * 8192);
  for (int i = tid; i < 16 * 256; i += 512) {
    int h = i >> 8, e2 = i & 255;
    float a0 = 0.f, a1 = 0.f;
    for (int nn = 0; nn < n; nn++) {
      float a = sS[h * SSTRIDE + nn];
      unsigned ew = ((const unsigned*)sEmb)[nn * 256 + e2];
      a0 += a * bfl(ew);
      a1 += a * bfh(ew);
    }
    U32[h * 256 + e2] = packbf2(a0, a1);
  }
}

// ---------------------------------------------------------------------------
// score = 10*tanh((mh . embed_n)/sqrt(E)) + ninf; softmax; first-index argmax
// ---------------------------------------------------------------------------
__global__ void __launch_bounds__(256) score_select(
    const u16* __restrict__ embB, int rowOff, int n,
    const float* __restrict__ mh, const float* __restrict__ ninf,
    int* __restrict__ sel, float* __restrict__ p) {
  __shared__ float sMh[512];
  __shared__ float sSc[160];
  __shared__ float red[256];
  __shared__ int redi[256];
  const int b = blockIdx.x, t = threadIdx.x;  // 256 threads
  for (int i = t; i < 512; i += 256) sMh[i] = mh[(long)b * 512 + i];
  __syncthreads();

  for (int nn = t; nn < n; nn += 256) {
    const unsigned* er = (const unsigned*)(embB + ((long)b * 160 + rowOff + nn) * 512);
    float acc = 0.f;
    for (int e2 = 0; e2 < 256; e2++) {
      unsigned w = er[e2];
      acc += bfl(w) * sMh[2 * e2] + bfh(w) * sMh[2 * e2 + 1];
    }
    sSc[nn] = 10.0f * tanhf(acc * 0.04419417382415922f) + ninf[(long)b * n + nn];
  }
  __syncthreads();

  float lm = NEG_INF;
  int li = 0x7fffffff;
  for (int nn = t; nn < n; nn += 256) {
    float v = sSc[nn];
    if (v > lm) { lm = v; li = nn; }
  }
  red[t] = lm; redi[t] = li;
  __syncthreads();
  for (int s = 128; s; s >>= 1) {
    if (t < s) {
      float v2 = red[t + s]; int i2 = redi[t + s];
      if (v2 > red[t] || (v2 == red[t] && i2 < redi[t])) { red[t] = v2; redi[t] = i2; }
    }
    __syncthreads();
  }
  float m = red[0];
  int am = redi[0];
  __syncthreads();

  float ls = 0.f;
  for (int nn = t; nn < n; nn += 256) ls += __expf(sSc[nn] - m);
  red[t] = ls;
  __syncthreads();
  for (int s = 128; s; s >>= 1) {
    if (t < s) red[t] += red[t + s];
    __syncthreads();
  }
  if (t == 0) {
    sel[b] = am;
    p[b] = 1.0f / red[0];
  }
}

// ---------------------------------------------------------------------------
// Final outputs: action (4 x B), log_prob (B), select_embed (B x 1536)
// ---------------------------------------------------------------------------
__global__ void __launch_bounds__(256) finalize(
    const float* __restrict__ embF,
    const int* __restrict__ opeStep, const int* __restrict__ endOpe,
    const int* __restrict__ selJob, const int* __restrict__ selMa,
    const int* __restrict__ selVeh,
    const float* __restrict__ pOpe, const float* __restrict__ pMa,
    const float* __restrict__ pVeh,
    const unsigned char* __restrict__ finish,
    float* __restrict__ out, int B) {
  const int b = blockIdx.x, t = threadIdx.x;
  const int sj = selJob[b], smm = selMa[b], sv = selVeh[b];
  if (t == 0) {
    int so = min(opeStep[(long)b * 100 + sj], endOpe[(long)b * 100 + sj]);
    out[0 * B + b] = (float)so;
    out[1 * B + b] = (float)smm;
    out[2 * B + b] = (float)sj;
    out[3 * B + b] = (float)sv;
    float po = pOpe[b], pm = pMa[b], pv = pVeh[b];
    if (finish[b]) { po = 1.f; pm = 1.f; pv = 1.f; }
    out[4 * B + b] = logf(po * pm * pv);
  }
  float* se = out + 5L * B + (long)b * 1536;
  const float* r0 = embF + ((long)b * 160 + sj) * 512;
  const float* r1 = embF + ((long)b * 160 + 100 + smm) * 512;
  const float* r2 = embF + ((long)b * 160 + 130 + sv) * 512;
  for (int i = t; i < 512; i += 256) {
    se[i] = r0[i];
    se[512 + i] = r1[i];
    se[1024 + i] = r2[i];
  }
}

// ---------------------------------------------------------------------------
// Host orchestration
// ---------------------------------------------------------------------------
extern "C" void kernel_launch(void* const* d_in, const int* in_sizes, int n_in,
                              void* d_out, int out_size, void* d_ws, size_t ws_size,
                              hipStream_t stream) {
  const int B = in_sizes[6];  // 1024
  const float RSQRT_QD = 0.17677669529663687f;  // 1/sqrt(32)

  const float* embF    = (const float*)d_in[0];
  const float* prev    = (const float*)d_in[1];
  const unsigned char* mask    = (const unsigned char*)d_in[2];
  const unsigned char* maskVeh = (const unsigned char*)d_in[3];
  const int* opeStep   = (const int*)d_in[4];
  const int* endOpe    = (const int*)d_in[5];
  const int* batchIdx  = (const int*)d_in[6];

  // --- workspace carve-up ---------------------------------------------------
  char* ws = (char*)d_ws;
  size_t off = 0;
  auto take = [&](size_t bytes) -> char* {
    char* r = ws + off;
    off = (off + bytes + 255) & ~(size_t)255;
    return r;
  };
  u16* embB   = (u16*)take((size_t)B * 160 * 512 * 2);
  u16* ctxA   = (u16*)take((size_t)B * 2048 * 2);
  u16* wB     = (u16*)take((size_t)(2048 * 512 + 16 * 512 * 512) * 2);
  u16* ctx0   = (u16*)take((size_t)B * 512 * 2);
  u16* qBuf   = (u16*)take((size_t)B * 512 * 2);
  u16* rBuf   = (u16*)take((size_t)B * 8192 * 2);
  u16* uBuf   = (u16*)take((size_t)B * 8192 * 2);
  u16* oBuf   = (u16*)take((size_t)B * 512 * 2);
  u16* ctxN   = (u16*)take((size_t)B * 512 * 2);
  float* mh   = (float*)take((size_t)B * 512 * 4);
  float* ninfN = (float*)take((size_t)B * 160 * 4);
  float* ninfO = (float*)take((size_t)B * 100 * 4);
  float* ninfV = (float*)take((size_t)B * 30 * 4);
  float* ninfM = (float*)take((size_t)B * 30 * 4);
  int* selJ = (int*)take((size_t)B * 4);
  int* selM = (int*)take((size_t)B * 4);
  int* selV = (int*)take((size_t)B * 4);
  float* pO = (float*)take((size_t)B * 4);
  float* pM = (float*)take((size_t)B * 4);
  float* pV = (float*)take((size_t)B * 4);
  unsigned char* fin = (unsigned char*)take((size_t)B);

  // bf16 weights: slot 0 = Wctx^T; 1-4 = Wq^T; 5-8 = Wk (untransposed: the
  // R-GEMM's K dim is the weight column -> already contiguous); 9-12 = Wv^T;
  // 13-16 = Wc^T.
  u16* wPtr[17];
  {
    size_t wo = 0;
    for (int i = 0; i < 17; i++) {
      wPtr[i] = wB + wo;
      wo += (i == 0) ? (size_t)2048 * 512 : (size_t)512 * 512;
    }
  }
  u16* wCtxT = wPtr[0];
  u16* wQT[4] = {wPtr[1], wPtr[2], wPtr[3], wPtr[4]};
  u16* wK[4]  = {wPtr[5], wPtr[6], wPtr[7], wPtr[8]};
  u16* wVT[4] = {wPtr[9], wPtr[10], wPtr[11], wPtr[12]};
  u16* wCT[4] = {wPtr[13], wPtr[14], wPtr[15], wPtr[16]};
  const float* bC[4] = {(const float*)d_in[22], (const float*)d_in[24],
                        (const float*)d_in[26], (const float*)d_in[28]};

  // --- phase 0: conversions, masks -----------------------------------------
  {
    // src index, transpose flag per slot
    const int srcIdx[17] = {7,  9, 10, 11, 12,  13, 15, 17, 19,  14, 16, 18, 20,  21, 23, 25, 27};
    const int trans[17]  = {1,  1, 1, 1, 1,     0, 0, 0, 0,      1, 1, 1, 1,      1, 1, 1, 1};
    ConvArgs ca;
    for (int i = 0; i < 17; i++) {
      ca.src[i] = (const float*)d_in[srcIdx[i]];
      ca.dst[i] = wPtr[i];
      ca.rows[i] = (i == 0) ? 2048 : 512;
      ca.cols[i] = 512;
      ca.trans[i] = trans[i];
    }
    convert_weights<<<dim3((2048 * 512 + 255) / 256, 17), 256, 0, stream>>>(ca);
  }
  mean_convert<<<B, 512, 0, stream>>>(embF, prev, embB, ctxA);
  ninf_precompute<<<B, 256, 0, stream>>>(mask, maskVeh, ninfN, ninfO, ninfV);
  finish_init<<<(B + 255) / 256, 256, 0, stream>>>(fin, B);
  finish_scatter<<<(in_sizes[6] + 255) / 256, 256, 0, stream>>>(batchIdx, fin, in_sizes[6]);

  auto gemm = [&](const u16* A, long lda, long aB_, const u16* Bp, long ldbn,
                  long bB_, u16* Cb, float* Cf, long ldc, long cB_,
                  const float* bias, int M, int N, int K, float alpha, int Z) {
    GemmArgs g;
    g.A = A; g.lda = lda; g.aBatch = aB_;
    g.B = Bp; g.ldbn = ldbn; g.bBatch = bB_;
    g.Cb = Cb; g.Cf = Cf; g.ldc = ldc; g.cBatch = cB_;
    g.bias = bias; g.M = M; g.N = N; g.K = K; g.alpha = alpha;
    wmma_gemm<<<dim3(M / 256, N / 32, Z), 256, 0, stream>>>(g);
  };

  hipFuncSetAttribute(reinterpret_cast<const void*>(attn_fused),
                      hipFuncAttributeMaxDynamicSharedMemorySize, 256 * 1024);

  // ctx0 = [graph, prev] @ Wctx + bctx  (M=B, K=2048, N=512)
  gemm(ctxA, 2048, 0, wCtxT, 2048, 0, ctx0, nullptr, 512, 0,
       (const float*)d_in[8], B, 512, 2048, 1.0f, 1);

  const int rowOffs[4] = {0, 0, 100, 130};
  const int ns[4] = {160, 100, 30, 30};
  const float* ninfs[4] = {ninfN, ninfO, ninfM, ninfV};

  for (int s = 0; s < 4; s++) {
    const u16* qsrc = (s == 0) ? ctx0 : ctxN;
    // q = qsrc @ Wq_s (M=B, K=512, N=512)
    gemm(qsrc, 512, 0, wQT[s], 512, 0, qBuf, nullptr, 512, 0, nullptr,
         B, 512, 512, 1.0f, 1);
    // R[b,h,e] = (1/sqrt(QD)) * sum_d q[b,h*32+d]*Wk[e,h*32+d]
    //   z=h: A=q (+32/head), B=wK original (b[n=e][k=d] = Wk[e*512+h*32+d])
    gemm(qBuf, 512, 32, wK[s], 512, 32, rBuf, nullptr, 8192, 512, nullptr,
         B, 512, 32, RSQRT_QD, 16);
    // fused softmax-attention (async LDS-staged embed slice)
    {
      int n = ns[s];
      size_t smem = (size_t)n * 512 * 2 + 16 * 520 * 2 + 16 * SSTRIDE * 4;
      attn_fused<<<B, 512, smem, stream>>>(embB, rowOffs[s], n, rBuf, ninfs[s], uBuf);
    }
    // O[b,h*32+d] = sum_e U[b,h,e]*Wv[e,h*32+d]
    //   z=h: A=U (+512/head), B=wV^T (b[n=d][k=e] = wVT[(h*32+d)*512+e])
    gemm(uBuf, 8192, 512, wVT[s], 512, 32 * 512, oBuf, nullptr, 512, 32, nullptr,
         B, 32, 512, 1.0f, 16);
    if (s == 0) {
      // context = O @ Wc_nodes + bc_nodes (bf16)
      gemm(oBuf, 512, 0, wCT[0], 512, 0, ctxN, nullptr, 512, 0, bC[0],
           B, 512, 512, 1.0f, 1);
    } else {
      // mh = O @ Wc_s + bc_s (fp32)
      gemm(oBuf, 512, 0, wCT[s], 512, 0, nullptr, mh, 512, 0, bC[s],
           B, 512, 512, 1.0f, 1);
      int* selp = (s == 1) ? selJ : (s == 2) ? selM : selV;
      float* pp = (s == 1) ? pO : (s == 2) ? pM : pV;
      score_select<<<B, 256, 0, stream>>>(embB, rowOffs[s], ns[s], mh, ninfs[s],
                                          selp, pp);
      if (s == 1) {
        ninf_ma_sel<<<(B * 30 + 255) / 256, 256, 0, stream>>>(mask, selJ, ninfM, B);
      }
    }
  }

  finalize<<<B, 256, 0, stream>>>(embF, opeStep, endOpe, selJ, selM, selV,
                                  pO, pM, pV, fin, (float*)d_out, B);
}
